// NTAttention_27908697490111
// MI455X (gfx1250) — matrix operand order repaired and checked by reading, not compiled, not run on hardware
//
#include <hip/hip_runtime.h>
#include <cstdint>

// Problem constants (fixed by the reference)
#define BB   128
#define LL   128
#define EMBD 1024
#define HH   8
#define DKK  128

typedef __attribute__((ext_vector_type(16))) __bf16 v16bf;
typedef __attribute__((ext_vector_type(8)))  float  v8f;
typedef __attribute__((ext_vector_type(4)))  unsigned int u32x4;
typedef __attribute__((ext_vector_type(8)))  int i32x8;
typedef __attribute__((ext_vector_type(4)))  int i32x4;

union AFrag { v16bf v; unsigned short u[16]; };
union F8    { v8f   v; float f[8]; };

__device__ __forceinline__ unsigned short f2bf(float f) {
  // round-to-nearest-even fp32 -> bf16
  unsigned int u = __float_as_uint(f);
  u += 0x7FFFu + ((u >> 16) & 1u);
  return (unsigned short)(u >> 16);
}

#define MODE_PLAIN 0
#define MODE_HEADS 1
#define MODE_RESID 2

#if __has_builtin(__builtin_amdgcn_tensor_load_to_lds) && \
    __has_builtin(__builtin_amdgcn_s_wait_tensorcnt)
#define USE_TDM 1
#else
#define USE_TDM 0
#endif

#if USE_TDM
// Issue one TDM tile DMA: 128 rows x 32 bf16 cols from a [dim1 x dim0] bf16
// matrix (row stride dim0 elements) into LDS with 8B padding after each 64B
// row (LDS row stride 72B == unsigned short[36]).
// D# packing per CDNA5 ISA.
// This toolchain exposes the 6-arg builtin:
//   (u32x4 g0, i32x8 g1, i32x4 g2, i32x4 g3, i32x8 extra, i32 cpol)
__device__ __forceinline__ void tdm_load_tile(unsigned lds_byte_addr,
                                              const unsigned short* gsrc,
                                              int dim0, int dim1) {
  const unsigned long long ga = (unsigned long long)(uintptr_t)gsrc;
  u32x4 g0;
  g0[0] = 1u;                                                // count=1 (valid)
  g0[1] = lds_byte_addr;                                     // lds_addr
  g0[2] = (unsigned)(ga & 0xFFFFFFFFull);                    // global_addr[31:0]
  g0[3] = (unsigned)((ga >> 32) & 0x1FFFFFFull) | (2u << 30);// addr[56:32]|type=2
  i32x8 g1;
  g1[0] = (1 << 16)        // data_size = 2 bytes
        | (1 << 20)        // pad_enable
        | (3 << 22)        // pad_interval: 16 DWORDs (one 64B tile row)
        | (1 << 25);       // pad_amount: 2 DWORDs (8B) -> 72B LDS row stride
  g1[1] = (dim0 & 0xFFFF) << 16;                             // tensor_dim0[15:0]
  g1[2] = ((dim0 >> 16) & 0xFFFF) | ((dim1 & 0xFFFF) << 16); // dim0[31:16]|dim1[15:0]
  g1[3] = ((dim1 >> 16) & 0xFFFF) | (32 << 16);              // dim1[31:16]|tile_dim0=32
  g1[4] = 128;                                               // tile_dim1=128, tile_dim2=0
  g1[5] = dim0;                                              // tensor_dim0_stride[31:0]
  g1[6] = 0;                                                 // stride[47:32], dim1_stride lo
  g1[7] = 0;
  i32x4 gz4 = {0, 0, 0, 0};                                  // 2D: groups 2/3 zero
  i32x8 gz8 = {0, 0, 0, 0, 0, 0, 0, 0};
  __builtin_amdgcn_tensor_load_to_lds(g0, g1, gz4, gz4, gz8, 0);
}
#endif

// ---------------------------------------------------------------------------
// C[M,N] = A[M,K] @ W[N,K]^T, A/W bf16, fp32 accumulate via v_wmma.
// Block tile 128x128, BK=32, 256 threads (8 waves), wave tile 32x64 (2x4 WMMA).
// TDM double-buffered LDS staging (wave 0 drives the DMA engine).
// ---------------------------------------------------------------------------
__global__ __launch_bounds__(256)
void gemm_bf16_wmma(const unsigned short* __restrict__ A,
                    const unsigned short* __restrict__ W,
                    float* __restrict__ C, const float* __restrict__ resid,
                    int M, int N, int K, int mode) {
  __shared__ unsigned short As[2][128][36];
  __shared__ unsigned short Bs[2][128][36];

  const int t    = threadIdx.x;
  const int lane = t & 31, wid = t >> 5;
  const int half = lane >> 4, lr = lane & 15;
  const int wm   = (wid & 3) * 32;    // 4 waves along M
  const int wn   = (wid >> 2) * 64;   // 2 waves along N
  const int m0   = blockIdx.y * 128, n0 = blockIdx.x * 128;

  v8f acc[2][4] = {};
  const int ksteps = K >> 5;

#if USE_TDM
  const unsigned ldsA[2] = {(unsigned)(uintptr_t)&As[0][0][0],
                            (unsigned)(uintptr_t)&As[1][0][0]};
  const unsigned ldsB[2] = {(unsigned)(uintptr_t)&Bs[0][0][0],
                            (unsigned)(uintptr_t)&Bs[1][0][0]};
  if (wid == 0) {  // prologue: tile 0 -> buffer 0
    tdm_load_tile(ldsA[0], A + (size_t)m0 * K, K, M);
    tdm_load_tile(ldsB[0], W + (size_t)n0 * K, K, N);
  }
#endif

  for (int kt = 0; kt < ksteps; ++kt) {
#if USE_TDM
    const int cur = kt & 1;
    if (wid == 0) {
      if (kt + 1 < ksteps) {  // issue next tile into other buffer, then wait
        const int k1 = (kt + 1) << 5;
        tdm_load_tile(ldsA[cur ^ 1], A + (size_t)m0 * K + k1, K, M);
        tdm_load_tile(ldsB[cur ^ 1], W + (size_t)n0 * K + k1, K, N);
        __builtin_amdgcn_s_wait_tensorcnt(2);  // tile kt done, kt+1 in flight
      } else {
        __builtin_amdgcn_s_wait_tensorcnt(0);
      }
    }
    __syncthreads();
#else
    const int cur = 0;
    const int k0 = kt << 5;
    #pragma unroll
    for (int i = 0; i < 2; ++i) {       // 512 x 16B chunks, 2 per thread
      const int idx = t + (i << 8);
      const int r = idx >> 2, c8 = (idx & 3) << 3;
      const uint4 av = *(const uint4*)(A + (size_t)(m0 + r) * K + k0 + c8);
      const uint4 bv = *(const uint4*)(W + (size_t)(n0 + r) * K + k0 + c8);
      *(uint2*)&As[0][r][c8]     = make_uint2(av.x, av.y);
      *(uint2*)&As[0][r][c8 + 4] = make_uint2(av.z, av.w);
      *(uint2*)&Bs[0][r][c8]     = make_uint2(bv.x, bv.y);
      *(uint2*)&Bs[0][r][c8 + 4] = make_uint2(bv.z, bv.w);
      if (kt + 1 < ksteps)
        __builtin_prefetch(A + (size_t)(m0 + r) * K + k0 + 32 + c8, 0, 1);
    }
    __syncthreads();
#endif

    const unsigned short (*as)[36] = As[cur];
    const unsigned short (*bs)[36] = Bs[cur];

    // Fragments per CDNA5 16-bit A (16x32) / B (32x16) VGPR layouts
    AFrag af[2], bfg[4];
    #pragma unroll
    for (int tm = 0; tm < 2; ++tm) {
      const int row = wm + tm * 16 + lr;
      #pragma unroll
      for (int i = 0; i < 8; ++i) {
        const int k = (i < 4 ? 2 * i : 16 + 2 * (i - 4)) + 8 * half;
        af[tm].u[2 * i]     = as[row][k];
        af[tm].u[2 * i + 1] = as[row][k + 1];
      }
    }
    #pragma unroll
    for (int tn = 0; tn < 4; ++tn) {
      const int col = wn + tn * 16 + lr;
      #pragma unroll
      for (int i = 0; i < 8; ++i) {
        const int k = (i < 4 ? 2 * i : 16 + 2 * (i - 4)) + 8 * half;
        bfg[tn].u[2 * i]     = bs[col][k];
        bfg[tn].u[2 * i + 1] = bs[col][k + 1];
      }
    }
    #pragma unroll
    for (int tm = 0; tm < 2; ++tm)
      #pragma unroll
      for (int tn = 0; tn < 4; ++tn)
        acc[tm][tn] = __builtin_amdgcn_wmma_f32_16x16x32_bf16(
            false, af[tm].v, false, bfg[tn].v, (short)0, acc[tm][tn],
            false, false);
    __syncthreads();
  }

  // Epilogue: C/D layout -> VGPR j holds M=j (lanes<16) / M=j+8 (lanes>=16)
  #pragma unroll
  for (int tm = 0; tm < 2; ++tm)
    #pragma unroll
    for (int tn = 0; tn < 4; ++tn) {
      F8 r; r.v = acc[tm][tn];
      #pragma unroll
      for (int j = 0; j < 8; ++j) {
        const int gr = m0 + wm + tm * 16 + half * 8 + j;
        const int gc = n0 + wn + tn * 16 + lr;
        const float val = r.f[j];
        if (mode == MODE_HEADS) {
          const int b = gr >> 7, l = gr & 127, h = gc >> 7, dk = gc & 127;
          C[((((size_t)b * HH + h) * LL + l) << 7) + dk] = val;
        } else if (mode == MODE_RESID) {
          const size_t o = (size_t)gr * N + gc;
          C[o] = val + resid[o];
        } else {
          C[(size_t)gr * N + gc] = val;
        }
      }
    }
}

// fp32 -> packed bf16 conversion prepass (8 elements/thread)
__global__ __launch_bounds__(256)
void cvt_bf16_kernel(const float* __restrict__ x, unsigned short* __restrict__ y) {
  const size_t o = ((size_t)blockIdx.x * 256 + threadIdx.x) * 8;
  const float4 a = *(const float4*)(x + o);
  const float4 b = *(const float4*)(x + o + 4);
  uint4 p;
  p.x = (unsigned)f2bf(a.x) | ((unsigned)f2bf(a.y) << 16);
  p.y = (unsigned)f2bf(a.z) | ((unsigned)f2bf(a.w) << 16);
  p.z = (unsigned)f2bf(b.x) | ((unsigned)f2bf(b.y) << 16);
  p.w = (unsigned)f2bf(b.z) | ((unsigned)f2bf(b.w) << 16);
  *(uint4*)(y + o) = p;
}

// PE table: f[h][r]; pe2d[h,r,c] = f[h][r]*f[h][c] (src_len == dk == 128)
__global__ __launch_bounds__(256)
void petab_kernel(float* __restrict__ pet) {
  const int tid = blockIdx.x * 256 + threadIdx.x;  // 1024
  const int h = tid >> 7, r = tid & 127;
  const float x = (float)(r & ~1);
  const float w = __expf(x * (-6.907755278982137f / 128.f));
  pet[tid] = (r & 1) ? __cosf((float)h * w) : __sinf((float)h * w);
}

// Wvs[dk][e] = sum_h W_V[h*DK+dk][e] in bf16 (ones-kernel V conv collapses heads)
__global__ __launch_bounds__(256)
void wsum_kernel(const float* __restrict__ WV, unsigned short* __restrict__ Wvs) {
  const int idx = blockIdx.x * 256 + threadIdx.x;  // 128*1024
  const int e = idx & 1023, dk = idx >> 10;
  float s = 0.f;
  #pragma unroll
  for (int h = 0; h < HH; ++h)
    s += WV[(((size_t)h * DKK + dk) << 10) + e];
  Wvs[idx] = f2bf(s);
}

// K conv: 3x3 SAME, 8->8 channels over (L, DK) per batch
__global__ __launch_bounds__(256)
void conv_k_kernel(const float* __restrict__ Kp, const float* __restrict__ kw,
                   const float* __restrict__ kb, float* __restrict__ Kc) {
  const int idx = blockIdx.x * 256 + threadIdx.x;  // B*H*L*DK
  const int dk = idx & 127, l = (idx >> 7) & 127;
  const int h = (idx >> 14) & 7, b = idx >> 17;
  float sum = kb[h];
  for (int hi = 0; hi < HH; ++hi) {
    const size_t base = ((size_t)(b * HH + hi)) << 14;
    #pragma unroll
    for (int dl = -1; dl <= 1; ++dl) {
      const int ll = l + dl;
      if (ll < 0 || ll > 127) continue;
      #pragma unroll
      for (int dd = -1; dd <= 1; ++dd) {
        const int cc = dk + dd;
        if (cc < 0 || cc > 127) continue;
        sum += Kp[base + (ll << 7) + cc] *
               kw[((h * HH + hi) * 3 + (dl + 1)) * 3 + (dd + 1)];
      }
    }
  }
  Kc[idx] = sum;
}

// V: 3x3 box-sum of Vsum (SAME padding); heads collapse because kernel is ones
__global__ __launch_bounds__(256)
void box_v_kernel(const float* __restrict__ Vsum, float* __restrict__ Vbox) {
  const int idx = blockIdx.x * 256 + threadIdx.x;  // B*L*DK
  const int dk = idx & 127, l = (idx >> 7) & 127, b = idx >> 14;
  const size_t base = ((size_t)b) << 14;
  float s = 0.f;
  #pragma unroll
  for (int dl = -1; dl <= 1; ++dl) {
    const int ll = l + dl;
    if (ll < 0 || ll > 127) continue;
    #pragma unroll
    for (int dd = -1; dd <= 1; ++dd) {
      const int cc = dk + dd;
      if (cc < 0 || cc > 127) continue;
      s += Vsum[base + (ll << 7) + cc];
    }
  }
  Vbox[idx] = s;
}

// Fused: score = Q*K/sqrt(DK), mask, softmax(128), +pe2d, context = attn*V
// One wave32 per 128-wide row (4 cols/lane), 8 rows per 256-thread block.
// ctx is written directly in bf16 for the FC WMMA GEMM.
__global__ __launch_bounds__(256)
void attn_kernel(const float* __restrict__ Qp, const float* __restrict__ Kc,
                 const float* __restrict__ Vb, const unsigned char* __restrict__ mask,
                 const float* __restrict__ pet, float* __restrict__ attn_out,
                 unsigned short* __restrict__ ctx) {
  const int lane = threadIdx.x & 31, wid = threadIdx.x >> 5;
  const int row = blockIdx.x * 8 + wid;       // (b*H + h)*L + l
  const int l = row & 127;
  const int h = (row >> 7) & 7;
  const int b = row >> 10;
  const size_t rbase = (size_t)row << 7;
  const size_t mbase = ((size_t)((b << 7) | l)) << 7;  // (b*L+l)*128
  const float scale = 0.08838834764831845f;            // 1/sqrt(128)

  float s[4];
  #pragma unroll
  for (int j = 0; j < 4; ++j) {
    const int c = lane + 32 * j;
    float sc = Qp[rbase + c] * Kc[rbase + c] * scale;
    if (mask[mbase + c]) sc = -1000.f;
    s[j] = sc;
  }
  float mx = fmaxf(fmaxf(s[0], s[1]), fmaxf(s[2], s[3]));
  #pragma unroll
  for (int o = 16; o >= 1; o >>= 1) mx = fmaxf(mx, __shfl_xor(mx, o, 32));
  float sum = 0.f;
  #pragma unroll
  for (int j = 0; j < 4; ++j) { s[j] = __expf(s[j] - mx); sum += s[j]; }
  #pragma unroll
  for (int o = 16; o >= 1; o >>= 1) sum += __shfl_xor(sum, o, 32);
  const float inv = 1.f / sum;
  const float fl = pet[(h << 7) + l];
  const size_t cbase = ((((size_t)((b << 7) | l)) << 3) | h) << 7;
  #pragma unroll
  for (int j = 0; j < 4; ++j) {
    const int c = lane + 32 * j;
    const float a = s[j] * inv + fl * pet[(h << 7) + c];
    attn_out[rbase + c] = a;
    ctx[cbase + c] = f2bf(a * Vb[mbase + c]);
  }
}

// LayerNorm over 1024, one 256-thread block per row
__global__ __launch_bounds__(256)
void ln_kernel(const float* __restrict__ x, const float* __restrict__ g,
               const float* __restrict__ be, float* __restrict__ out) {
  const int row = blockIdx.x;
  const int t = threadIdx.x;
  const size_t base = (size_t)row << 10;
  float v[4], s = 0.f, s2 = 0.f;
  #pragma unroll
  for (int j = 0; j < 4; ++j) {
    v[j] = x[base + t + 256 * j];
    s += v[j]; s2 += v[j] * v[j];
  }
  #pragma unroll
  for (int o = 16; o >= 1; o >>= 1) {
    s  += __shfl_xor(s, o, 32);
    s2 += __shfl_xor(s2, o, 32);
  }
  __shared__ float r1[8], r2[8];
  if ((t & 31) == 0) { r1[t >> 5] = s; r2[t >> 5] = s2; }
  __syncthreads();
  s = 0.f; s2 = 0.f;
  #pragma unroll
  for (int i = 0; i < 8; ++i) { s += r1[i]; s2 += r2[i]; }
  const float mean = s * (1.f / 1024.f);
  const float var  = s2 * (1.f / 1024.f) - mean * mean;
  const float rstd = rsqrtf(var + 1e-5f);
  #pragma unroll
  for (int j = 0; j < 4; ++j) {
    const int c = t + 256 * j;
    out[base + c] = (v[j] - mean) * rstd * g[c] + be[c];
  }
}

extern "C" void kernel_launch(void* const* d_in, const int* in_sizes, int n_in,
                              void* d_out, int out_size, void* d_ws, size_t ws_size,
                              hipStream_t stream) {
  (void)in_sizes; (void)n_in; (void)out_size; (void)ws_size;
  const float* inQ  = (const float*)d_in[0];
  const float* inK  = (const float*)d_in[1];
  const float* inV  = (const float*)d_in[2];
  const unsigned char* mask = (const unsigned char*)d_in[3];
  const float* WQ   = (const float*)d_in[4];
  const float* WK   = (const float*)d_in[5];
  const float* WV   = (const float*)d_in[6];
  const float* kw   = (const float*)d_in[7];
  const float* kb   = (const float*)d_in[8];
  const float* fcw  = (const float*)d_in[9];
  const float* gam  = (const float*)d_in[10];
  const float* bet  = (const float*)d_in[11];

  // Workspace layout (bytes); ~325 MB total
  char* base = (char*)d_ws;
  float* Qp    = (float*)(base + 0);            // [B,H,L,DK] f32   64MB
  float* Kp    = (float*)(base + 67108864);     // [B,H,L,DK] f32   (reused preLN)
  float* Kc    = (float*)(base + 134217728);    // [B,H,L,DK] f32
  float* Vsum  = (float*)(base + 201326592);    // [B,L,DK]   f32    8MB
  float* Vbox  = (float*)(base + 209715200);    // [B,L,DK]   f32
  float* pet   = (float*)(base + 218103808);    // [H,128]    f32    4KB
  unsigned short* inQbf = (unsigned short*)(base + 218107904);  // 32MB (reused ctxbf)
  unsigned short* inKbf = (unsigned short*)(base + 251662336);  // 32MB
  unsigned short* inVbf = (unsigned short*)(base + 285216768);  // 32MB
  unsigned short* WQbf  = (unsigned short*)(base + 318771200);  // 2MB
  unsigned short* WKbf  = (unsigned short*)(base + 320868352);  // 2MB
  unsigned short* fcwbf = (unsigned short*)(base + 322965504);  // 2MB
  unsigned short* Wvsbf = (unsigned short*)(base + 325062656);  // 256KB
  float* preLN = Kp;                    // Kp dead after conv_k
  unsigned short* ctxbf = inQbf;        // inQbf dead after Q projection

  float* outLN   = (float*)d_out;                      // [B,L,EMB]
  float* outAttn = outLN + (size_t)BB * LL * EMBD;     // [B,H,L,DK]

  // Prepass: fp32 -> bf16 conversions, PE table, head-collapsed W_V
  cvt_bf16_kernel<<<8192, 256, 0, stream>>>(inQ, inQbf);
  cvt_bf16_kernel<<<8192, 256, 0, stream>>>(inK, inKbf);
  cvt_bf16_kernel<<<8192, 256, 0, stream>>>(inV, inVbf);
  cvt_bf16_kernel<<<512, 256, 0, stream>>>(WQ, WQbf);
  cvt_bf16_kernel<<<512, 256, 0, stream>>>(WK, WKbf);
  cvt_bf16_kernel<<<512, 256, 0, stream>>>(fcw, fcwbf);
  petab_kernel<<<4, 256, 0, stream>>>(pet);
  wsum_kernel<<<512, 256, 0, stream>>>(WV, Wvsbf);

  // Projections (bf16 WMMA GEMMs with TDM-staged LDS tiles)
  gemm_bf16_wmma<<<dim3(8, 128), 256, 0, stream>>>(inQbf, WQbf, Qp, nullptr,
                                                   16384, 1024, 1024, MODE_HEADS);
  gemm_bf16_wmma<<<dim3(8, 128), 256, 0, stream>>>(inKbf, WKbf, Kp, nullptr,
                                                   16384, 1024, 1024, MODE_HEADS);
  gemm_bf16_wmma<<<dim3(1, 128), 256, 0, stream>>>(inVbf, Wvsbf, Vsum, nullptr,
                                                   16384, 128, 1024, MODE_PLAIN);

  conv_k_kernel<<<65536, 256, 0, stream>>>(Kp, kw, kb, Kc);
  box_v_kernel<<<8192, 256, 0, stream>>>(Vsum, Vbox);
  attn_kernel<<<16384, 256, 0, stream>>>(Qp, Kc, Vbox, mask, pet, outAttn, ctxbf);

  // FC + residual (bf16 WMMA GEMM), then LayerNorm
  gemm_bf16_wmma<<<dim3(8, 128), 256, 0, stream>>>(ctxbf, fcwbf, preLN, inQ,
                                                   16384, 1024, 1024, MODE_RESID);
  ln_kernel<<<16384, 256, 0, stream>>>(preLN, gam, bet, outLN);
}
